// SharedTransformer_24378234372240
// MI455X (gfx1250) — matrix operand order repaired and checked by reading, not compiled, not run on hardware
//
#include <hip/hip_runtime.h>
#include <hip/hip_bf16.h>

// ---------------------------------------------------------------------------
// MI455X (gfx1250) implementation of the PCT offset-attention block.
// All matmuls run on v_wmma_f32_16x16x32_f16 (f16 in, f32 accumulate).
// The dominant GEMM (x_v @ attention, K=2048) uses a cooperative workgroup
// tile with double-buffered GLOBAL_LOAD_ASYNC_TO_LDS_B128 staging of the
// attention panel (ASYNCcnt-tracked, s_wait_asynccnt synchronized).
// B=8, Cin=64, C=256, N=2048.  Workspace usage: ~117 MB.
// ---------------------------------------------------------------------------

typedef __attribute__((ext_vector_type(16))) _Float16 v16h;
typedef __attribute__((ext_vector_type(8)))  _Float16 v8h;
typedef __attribute__((ext_vector_type(8)))  float    v8f;

#define B_   8
#define CIN_ 64
#define C_   256
#define N_   2048

// ---------------------------------------------------------------------------
// WMMA core helpers
// ---------------------------------------------------------------------------
__device__ __forceinline__ v8f wmma_f16(v16h a, v16h b, v8f c) {
  // D = A(16x32 f16) * B(32x16 f16) + C(16x16 f32)
  return __builtin_amdgcn_wmma_f32_16x16x32_f16(
      /*neg_a=*/false, a, /*neg_b=*/false, b,
      /*c_mod=*/(short)0, c, /*reuse_a=*/false, /*reuse_b=*/false);
}

// A fragment: row-major [M,K] f16. Lane L holds row (L&15); lanes 0-15 take
// K = k0+0..7 and k0+16..23, lanes 16-31 take K = k0+8..15 and k0+24..31.
// `p` already points at A + m*lda + k0 + ((lane>>4)<<3).
__device__ __forceinline__ v16h ldA(const _Float16* p) {
  union { v16h v; v8h h[2]; } u;
  u.h[0] = *(const v8h*)(p);
  u.h[1] = *(const v8h*)(p + 16);
  return u.v;
}

// B fragment from Bt layout [Ncols, K] row-major. Lane L holds column (L&15);
// lanes 0-15 take K = k0+0..7 and k0+16..23, lanes 16-31 take K = k0+8..15 and k0+24..31.
// `p` already points at Bt + n*ldb + k0 + ((lane>>4)<<3).
__device__ __forceinline__ v16h ldB(const _Float16* p) {
  union { v16h v; v8h h[2]; } u;
  u.h[0] = *(const v8h*)(p);
  u.h[1] = *(const v8h*)(p + 16);
  return u.v;
}

// One wave computes a 16(M) x 64(N) f32 tile; A frag reused across 4 B frags.
__device__ __forceinline__ void wave_gemm_16x64(
    const _Float16* __restrict__ A, int lda,
    const _Float16* __restrict__ Bt, int ldb,
    int m0, int n0, int K, v8f acc[4]) {
  const int lane = threadIdx.x & 31;
  const int hw   = lane >> 4;
  const _Float16* arow = A  + (size_t)(m0 + (lane & 15)) * lda + (hw << 3);
  const _Float16* brow = Bt + (size_t)(n0 + (lane & 15)) * ldb + (hw << 3);
  for (int k = 0; k < K; k += 32) {
    v16h a = ldA(arow + k);
    __builtin_prefetch(arow + k + 128, 0, 1);   // global_prefetch_b8
#pragma unroll
    for (int j = 0; j < 4; ++j) {
      v16h b = ldB(brow + (size_t)(j * 16) * ldb + k);
      acc[j] = wmma_f16(a, b, acc[j]);
    }
  }
}

__device__ __forceinline__ void zero_acc(v8f acc[4]) {
#pragma unroll
  for (int j = 0; j < 4; ++j)
#pragma unroll
    for (int r = 0; r < 8; ++r) acc[j][r] = 0.0f;
}

// Decode (batch, m-tile, 64-wide n-group) from global wave id.
__device__ __forceinline__ void tile_decode(int tiles_m, int& b, int& mt, int& ng) {
  const int tiles_ng = N_ / 64;
  int tid = blockIdx.x * (blockDim.x >> 5) + (threadIdx.x >> 5);
  ng = tid % tiles_ng;
  int t2 = tid / tiles_ng;
  mt = t2 % tiles_m;
  b  = t2 / tiles_m;
}

// ---------------------------------------------------------------------------
// CDNA5 async global->LDS copy (ASYNCcnt tracked).  One b128 per lane.
// VDST = VGPR holding the LDS byte address; VADDR = 64-bit global address.
// Generic pointers to __shared__ carry the LDS offset in their low 32 bits.
// ---------------------------------------------------------------------------
__device__ __forceinline__ void async_ld_b128(void* lds, const void* gsrc) {
  unsigned lds_off = (unsigned)(size_t)lds;
  asm volatile("global_load_async_to_lds_b128 %0, %1, off"
               :: "v"(lds_off), "v"(gsrc) : "memory");
}
__device__ __forceinline__ void wait_async_le1() {
  asm volatile("s_wait_asynccnt 0x1" ::: "memory");
}
__device__ __forceinline__ void wait_async_0() {
  asm volatile("s_wait_asynccnt 0x0" ::: "memory");
}

// ---------------------------------------------------------------------------
// Prep kernels
// ---------------------------------------------------------------------------
__global__ void k_cvt_f16(const float* __restrict__ src, _Float16* __restrict__ dst, int n) {
  int i = blockIdx.x * blockDim.x + threadIdx.x;
  if (i < n) dst[i] = (_Float16)src[i];
}

__global__ void k_bn_prep(const float* __restrict__ g, const float* __restrict__ b,
                          const float* __restrict__ m, const float* __restrict__ v,
                          float* __restrict__ scale, float* __restrict__ shift, int n) {
  int i = blockIdx.x * blockDim.x + threadIdx.x;
  if (i < n) {
    float s = g[i] * rsqrtf(v[i] + 1e-5f);
    scale[i] = s;
    shift[i] = b[i] - m[i] * s;
  }
}

// inputs [B,Cin,N] f32 -> inT [B,N,Cin] f16 (B-operand layout for the stem GEMM)
__global__ void k_transpose_in(const float* __restrict__ src, _Float16* __restrict__ dst) {
  int i = blockIdx.x * blockDim.x + threadIdx.x;   // B*Cin*N threads
  int n = i % N_;
  int c = (i / N_) % CIN_;
  int b = i / (N_ * CIN_);
  dst[((size_t)b * N_ + n) * CIN_ + c] = (_Float16)src[i];
}

__global__ void k_zero(float* __restrict__ p, int n) {
  int i = blockIdx.x * blockDim.x + threadIdx.x;
  if (i < n) p[i] = 0.0f;
}

// ---------------------------------------------------------------------------
// Stem: x = relu(bn1(conv1_w @ inputs));  writes x f32 [B,C,N] and
// x f16 into xcatT[:, :, 0:256]  (transposed [B,N,512] concat buffer)
// ---------------------------------------------------------------------------
__global__ void k_stem(const _Float16* __restrict__ W1, const _Float16* __restrict__ inT,
                       const float* __restrict__ s1, const float* __restrict__ sh1,
                       float* __restrict__ x, _Float16* __restrict__ xcatT) {
  int b, mt, ng; tile_decode(C_ / 16, b, mt, ng);
  v8f acc[4]; zero_acc(acc);
  wave_gemm_16x64(W1, CIN_, inT + (size_t)b * N_ * CIN_, CIN_, mt * 16, ng * 64, CIN_, acc);
  const int lane = threadIdx.x & 31;
  const int ro = (lane >> 4) << 3, cl = lane & 15;
#pragma unroll
  for (int j = 0; j < 4; ++j) {
    int n = ng * 64 + j * 16 + cl;
#pragma unroll
    for (int r = 0; r < 8; ++r) {
      int m = mt * 16 + ro + r;
      float v = fmaxf(acc[j][r] * s1[m] + sh1[m], 0.0f);
      x[((size_t)b * C_ + m) * N_ + n] = v;
      xcatT[((size_t)b * N_ + n) * (2 * C_) + m] = (_Float16)v;
    }
  }
}

// ---------------------------------------------------------------------------
// x_qk = wqk @ x  -> stored transposed [B,N,64] f16 (serves as BOTH A and Bt
// operands of the energy GEMM)
// ---------------------------------------------------------------------------
__global__ void k_qk(const _Float16* __restrict__ Wqk, const _Float16* __restrict__ xcatT,
                     _Float16* __restrict__ xqkT) {
  int b, mt, ng; tile_decode(CIN_ / 16, b, mt, ng);
  v8f acc[4]; zero_acc(acc);
  wave_gemm_16x64(Wqk, C_, xcatT + (size_t)b * N_ * (2 * C_), 2 * C_,
                  mt * 16, ng * 64, C_, acc);
  const int lane = threadIdx.x & 31;
  const int ro = (lane >> 4) << 3, cl = lane & 15;
#pragma unroll
  for (int j = 0; j < 4; ++j) {
    int n = ng * 64 + j * 16 + cl;
#pragma unroll
    for (int r = 0; r < 8; ++r) {
      int m = mt * 16 + ro + r;
      xqkT[((size_t)b * N_ + n) * CIN_ + m] = (_Float16)acc[j][r];
    }
  }
}

// ---------------------------------------------------------------------------
// x_v = v_w @ x + v_b -> f16 row-major [B,C,N] (A-operand of the x_r GEMM)
// ---------------------------------------------------------------------------
__global__ void k_v(const _Float16* __restrict__ Vw, const _Float16* __restrict__ xcatT,
                    const float* __restrict__ vb, _Float16* __restrict__ xv) {
  int b, mt, ng; tile_decode(C_ / 16, b, mt, ng);
  v8f acc[4]; zero_acc(acc);
  wave_gemm_16x64(Vw, C_, xcatT + (size_t)b * N_ * (2 * C_), 2 * C_,
                  mt * 16, ng * 64, C_, acc);
  const int lane = threadIdx.x & 31;
  const int ro = (lane >> 4) << 3, cl = lane & 15;
#pragma unroll
  for (int j = 0; j < 4; ++j) {
    int n = ng * 64 + j * 16 + cl;
#pragma unroll
    for (int r = 0; r < 8; ++r) {
      int m = mt * 16 + ro + r;
      xv[((size_t)b * C_ + m) * N_ + n] = (_Float16)(acc[j][r] + vb[m]);
    }
  }
}

// ---------------------------------------------------------------------------
// Attention: one workgroup owns a 16-row strip of energy[n0:n0+16, 0:2048].
// Phase 1: 8 waves compute qk^T qk tiles via WMMA (+rel_pos) into 128 KB LDS.
// Phase 2: row softmax (wave32 shfl reductions), write attn^T f16 [B,N(m),N(n)]
// and atomically accumulate colsum[m] = sum_n attn[n,m] for the L1 renorm.
// ---------------------------------------------------------------------------
__global__ void k_attn(const _Float16* __restrict__ xqkT, const float* __restrict__ rel_pos,
                       _Float16* __restrict__ attnT, float* __restrict__ colsum) {
  extern __shared__ float sE[];                 // 16 * 2048 f32 = 128 KB
  const int b    = blockIdx.x / (N_ / 16);
  const int n0   = (blockIdx.x % (N_ / 16)) * 16;
  const int wave = threadIdx.x >> 5;
  const int lane = threadIdx.x & 31;
  const int ro = (lane >> 4) << 3, cl = lane & 15;
  const _Float16* Q = xqkT + (size_t)b * N_ * CIN_;

  for (int g = wave; g < N_ / 64; g += 8) {
    v8f acc[4]; zero_acc(acc);
    wave_gemm_16x64(Q, CIN_, Q, CIN_, n0, g * 64, CIN_, acc);
#pragma unroll
    for (int j = 0; j < 4; ++j) {
      int m = g * 64 + j * 16 + cl;
#pragma unroll
      for (int r = 0; r < 8; ++r) {
        int nl = ro + r;
        float e = acc[j][r] + rel_pos[((size_t)b * N_ + (n0 + nl)) * N_ + m];
        sE[nl * N_ + m] = e;
      }
    }
  }
  __syncthreads();

  // each wave softmaxes rows 2*wave and 2*wave+1
  const int nl0 = wave * 2, nl1 = nl0 + 1;
  float mx0 = -3.402823466e38f, mx1 = -3.402823466e38f;
  for (int t = lane; t < N_; t += 32) {
    mx0 = fmaxf(mx0, sE[nl0 * N_ + t]);
    mx1 = fmaxf(mx1, sE[nl1 * N_ + t]);
  }
#pragma unroll
  for (int o = 16; o > 0; o >>= 1) {
    mx0 = fmaxf(mx0, __shfl_xor(mx0, o, 32));
    mx1 = fmaxf(mx1, __shfl_xor(mx1, o, 32));
  }
  float s0 = 0.0f, s1 = 0.0f;
  for (int t = lane; t < N_; t += 32) {
    float e0 = __expf(sE[nl0 * N_ + t] - mx0);
    float e1 = __expf(sE[nl1 * N_ + t] - mx1);
    sE[nl0 * N_ + t] = e0;  s0 += e0;
    sE[nl1 * N_ + t] = e1;  s1 += e1;
  }
#pragma unroll
  for (int o = 16; o > 0; o >>= 1) {
    s0 += __shfl_xor(s0, o, 32);
    s1 += __shfl_xor(s1, o, 32);
  }
  const float i0 = 1.0f / s0, i1 = 1.0f / s1;
  const int ng0 = n0 + nl0, ng1 = n0 + nl1;
  for (int t = lane; t < N_; t += 32) {
    float a0 = sE[nl0 * N_ + t] * i0;
    float a1 = sE[nl1 * N_ + t] * i1;
    attnT[((size_t)b * N_ + t) * N_ + ng0] = (_Float16)a0;
    attnT[((size_t)b * N_ + t) * N_ + ng1] = (_Float16)a1;
    atomicAdd(&colsum[b * N_ + t], a0 + a1);
  }
}

// ---------------------------------------------------------------------------
// x_r = (x_v @ attn) / (1e-9 + colsum[m]);  t = x - x_r  -> tT f16 [B,N,C]
// Cooperative 128x64 block tile, K = 2048.  The 64x32 attention B-panel per
// k-step is staged in LDS via double-buffered async global->LDS b128 copies
// (one per thread), so all 8 waves share each B element.  Rows padded to 40
// halves (20 banks) so B-fragment reads are bank-conflict free.
// ---------------------------------------------------------------------------
__global__ void __launch_bounds__(256)
k_xr(const _Float16* __restrict__ xv, const _Float16* __restrict__ attnT,
     const float* __restrict__ colsum, const float* __restrict__ x,
     _Float16* __restrict__ tT) {
  __shared__ _Float16 sB[2][64][40];
  const int tid  = threadIdx.x;
  const int wave = tid >> 5;
  const int lane = tid & 31;
  // grid: b(8) * mblock(2) * ngroup(32)
  const int ng = blockIdx.x & 31;
  const int mb = (blockIdx.x >> 5) & 1;
  const int b  = blockIdx.x >> 6;
  const int n0 = ng * 64;
  const int m0 = mb * 128 + wave * 16;

  const _Float16* A  = xv    + (size_t)b * C_ * N_;
  const _Float16* Bt = attnT + (size_t)b * N_ * N_;

  // each thread owns one 16B chunk of the 64x32 B tile
  const int brow = tid >> 2;            // 0..63
  const int bch  = (tid & 3) * 8;       // 0,8,16,24 halves
  const _Float16* gsrc = Bt + (size_t)(n0 + brow) * N_ + bch;

  const int hw = lane >> 4;
  const int ro = (hw << 3), cl = lane & 15;
  const _Float16* arow = A + (size_t)(m0 + cl) * N_ + (hw << 3);

  v8f acc[4]; zero_acc(acc);

  // prologue: k-steps 0 and 1 in flight
  async_ld_b128(&sB[0][brow][bch], gsrc);
  async_ld_b128(&sB[1][brow][bch], gsrc + 32);

  for (int ks = 0; ks < N_ / 32; ++ks) {
    if (ks + 1 < N_ / 32) wait_async_le1();   // oldest batch (this buffer) done
    else                  wait_async_0();
    __syncthreads();
    const _Float16* bbase = &sB[ks & 1][0][0];
    v16h a = ldA(arow + ks * 32);
#pragma unroll
    for (int j = 0; j < 4; ++j) {
      v16h bf = ldB(bbase + (j * 16 + cl) * 40 + (hw << 3));
      acc[j] = wmma_f16(a, bf, acc[j]);
    }
    __syncthreads();                           // all waves done reading buffer
    if (ks + 2 < N_ / 32)
      async_ld_b128(&sB[ks & 1][brow][bch], gsrc + (ks + 2) * 32);
  }

#pragma unroll
  for (int j = 0; j < 4; ++j) {
    int m = n0 + j * 16 + cl;                  // attention output column
    float inv = 1.0f / (1e-9f + colsum[b * N_ + m]);
#pragma unroll
    for (int r = 0; r < 8; ++r) {
      int c = m0 + ro + r;
      float xr = acc[j][r] * inv;
      float t = x[((size_t)b * C_ + c) * N_ + m] - xr;
      tT[((size_t)b * N_ + m) * C_ + c] = (_Float16)t;
    }
  }
}

// ---------------------------------------------------------------------------
// x1 = x + relu(bn_an(trans_w @ t + trans_b)) -> f16 into xcatT[:, :, 256:512]
// ---------------------------------------------------------------------------
__global__ void k_trans(const _Float16* __restrict__ Tw, const _Float16* __restrict__ tT,
                        const float* __restrict__ tb, const float* __restrict__ s_an,
                        const float* __restrict__ sh_an, const float* __restrict__ x,
                        _Float16* __restrict__ xcatT) {
  int b, mt, ng; tile_decode(C_ / 16, b, mt, ng);
  v8f acc[4]; zero_acc(acc);
  wave_gemm_16x64(Tw, C_, tT + (size_t)b * N_ * C_, C_, mt * 16, ng * 64, C_, acc);
  const int lane = threadIdx.x & 31;
  const int ro = (lane >> 4) << 3, cl = lane & 15;
#pragma unroll
  for (int j = 0; j < 4; ++j) {
    int n = ng * 64 + j * 16 + cl;
#pragma unroll
    for (int r = 0; r < 8; ++r) {
      int c = mt * 16 + ro + r;
      float u = acc[j][r] + tb[c];
      u = fmaxf(u * s_an[c] + sh_an[c], 0.0f);
      float x1 = x[((size_t)b * C_ + c) * N_ + n] + u;
      xcatT[((size_t)b * N_ + n) * (2 * C_) + C_ + c] = (_Float16)x1;
    }
  }
}

// ---------------------------------------------------------------------------
// out = leaky_relu(bn_f(fuse_w @ concat(x, x1)), 0.2)   K = 512, f32 output
// ---------------------------------------------------------------------------
__global__ void k_fuse(const _Float16* __restrict__ Fw, const _Float16* __restrict__ xcatT,
                       const float* __restrict__ s_f, const float* __restrict__ sh_f,
                       float* __restrict__ out) {
  int b, mt, ng; tile_decode(C_ / 16, b, mt, ng);
  v8f acc[4]; zero_acc(acc);
  wave_gemm_16x64(Fw, 2 * C_, xcatT + (size_t)b * N_ * (2 * C_), 2 * C_,
                  mt * 16, ng * 64, 2 * C_, acc);
  const int lane = threadIdx.x & 31;
  const int ro = (lane >> 4) << 3, cl = lane & 15;
#pragma unroll
  for (int j = 0; j < 4; ++j) {
    int n = ng * 64 + j * 16 + cl;
#pragma unroll
    for (int r = 0; r < 8; ++r) {
      int c = mt * 16 + ro + r;
      float o = acc[j][r] * s_f[c] + sh_f[c];
      o = (o < 0.0f) ? 0.2f * o : o;
      out[((size_t)b * C_ + c) * N_ + n] = o;
    }
  }
}

// ---------------------------------------------------------------------------
// Host launcher
// ---------------------------------------------------------------------------
extern "C" void kernel_launch(void* const* d_in, const int* in_sizes, int n_in,
                              void* d_out, int out_size, void* d_ws, size_t ws_size,
                              hipStream_t stream) {
  const float* inputs  = (const float*)d_in[0];
  const float* rel_pos = (const float*)d_in[1];
  const float* conv1_w = (const float*)d_in[2];
  const float* bn1_g   = (const float*)d_in[3];
  const float* bn1_b   = (const float*)d_in[4];
  const float* bn1_m   = (const float*)d_in[5];
  const float* bn1_v   = (const float*)d_in[6];
  const float* wqk     = (const float*)d_in[7];
  const float* v_w     = (const float*)d_in[8];
  const float* v_b     = (const float*)d_in[9];
  const float* trans_w = (const float*)d_in[10];
  const float* trans_b = (const float*)d_in[11];
  const float* an_g    = (const float*)d_in[12];
  const float* an_b    = (const float*)d_in[13];
  const float* an_m    = (const float*)d_in[14];
  const float* an_v    = (const float*)d_in[15];
  const float* fuse_w  = (const float*)d_in[16];
  const float* fbn_g   = (const float*)d_in[17];
  const float* fbn_b   = (const float*)d_in[18];
  const float* fbn_m   = (const float*)d_in[19];
  const float* fbn_v   = (const float*)d_in[20];
  float* out = (float*)d_out;

  char* ws = (char*)d_ws;
  // ---- workspace layout (bytes) ----
  const size_t OFF_X     = 0;                       // x f32   [B,C,N]   16 MB
  const size_t OFF_XCATT = OFF_X     + (size_t)B_ * C_ * N_ * 4;        // [B,N,512] f16  16 MB
  const size_t OFF_INT   = OFF_XCATT + (size_t)B_ * N_ * 2 * C_ * 2;    // [B,N,64]  f16   2 MB
  const size_t OFF_XQKT  = OFF_INT   + (size_t)B_ * N_ * CIN_ * 2;      // [B,N,64]  f16   2 MB
  const size_t OFF_XV    = OFF_XQKT  + (size_t)B_ * N_ * CIN_ * 2;      // [B,C,N]   f16   8 MB
  const size_t OFF_ATTNT = OFF_XV    + (size_t)B_ * C_ * N_ * 2;        // [B,N,N]   f16  64 MB
  const size_t OFF_TT    = OFF_ATTNT + (size_t)B_ * N_ * N_ * 2;        // [B,N,C]   f16   8 MB
  const size_t OFF_CS    = OFF_TT    + (size_t)B_ * N_ * C_ * 2;        // colsum f32 64 KB
  const size_t OFF_W1H   = OFF_CS    + (size_t)B_ * N_ * 4;
  const size_t OFF_WQKH  = OFF_W1H   + (size_t)C_ * CIN_ * 2;
  const size_t OFF_VWH   = OFF_WQKH  + (size_t)CIN_ * C_ * 2;
  const size_t OFF_TWH   = OFF_VWH   + (size_t)C_ * C_ * 2;
  const size_t OFF_FWH   = OFF_TWH   + (size_t)C_ * C_ * 2;
  const size_t OFF_S1    = OFF_FWH   + (size_t)C_ * 2 * C_ * 2;
  const size_t OFF_SH1   = OFF_S1    + C_ * 4;
  const size_t OFF_SAN   = OFF_SH1   + C_ * 4;
  const size_t OFF_SHAN  = OFF_SAN   + C_ * 4;
  const size_t OFF_SF    = OFF_SHAN  + C_ * 4;
  const size_t OFF_SHF   = OFF_SF    + C_ * 4;

  _Float16* w1h   = (_Float16*)(ws + OFF_W1H);
  _Float16* wqkh  = (_Float16*)(ws + OFF_WQKH);
  _Float16* vwh   = (_Float16*)(ws + OFF_VWH);
  _Float16* twh   = (_Float16*)(ws + OFF_TWH);
  _Float16* fwh   = (_Float16*)(ws + OFF_FWH);
  _Float16* inT   = (_Float16*)(ws + OFF_INT);
  _Float16* xcatT = (_Float16*)(ws + OFF_XCATT);
  _Float16* xqkT  = (_Float16*)(ws + OFF_XQKT);
  _Float16* xv    = (_Float16*)(ws + OFF_XV);
  _Float16* attnT = (_Float16*)(ws + OFF_ATTNT);
  _Float16* tT    = (_Float16*)(ws + OFF_TT);
  float* x      = (float*)(ws + OFF_X);
  float* colsum = (float*)(ws + OFF_CS);
  float* s1  = (float*)(ws + OFF_S1);   float* sh1  = (float*)(ws + OFF_SH1);
  float* san = (float*)(ws + OFF_SAN);  float* shan = (float*)(ws + OFF_SHAN);
  float* sf  = (float*)(ws + OFF_SF);   float* shf  = (float*)(ws + OFF_SHF);

  // ---- prep ----
  k_cvt_f16<<<(C_ * CIN_ + 255) / 256, 256, 0, stream>>>(conv1_w, w1h, C_ * CIN_);
  k_cvt_f16<<<(CIN_ * C_ + 255) / 256, 256, 0, stream>>>(wqk, wqkh, CIN_ * C_);
  k_cvt_f16<<<(C_ * C_ + 255) / 256, 256, 0, stream>>>(v_w, vwh, C_ * C_);
  k_cvt_f16<<<(C_ * C_ + 255) / 256, 256, 0, stream>>>(trans_w, twh, C_ * C_);
  k_cvt_f16<<<(C_ * 2 * C_ + 255) / 256, 256, 0, stream>>>(fuse_w, fwh, C_ * 2 * C_);
  k_bn_prep<<<1, 256, 0, stream>>>(bn1_g, bn1_b, bn1_m, bn1_v, s1, sh1, C_);
  k_bn_prep<<<1, 256, 0, stream>>>(an_g, an_b, an_m, an_v, san, shan, C_);
  k_bn_prep<<<1, 256, 0, stream>>>(fbn_g, fbn_b, fbn_m, fbn_v, sf, shf, C_);
  k_transpose_in<<<(B_ * CIN_ * N_) / 256, 256, 0, stream>>>(inputs, inT);
  k_zero<<<(B_ * N_) / 256, 256, 0, stream>>>(colsum, B_ * N_);

  // ---- pipeline (wave tiles = B * (M/16) * (N/64), 8 waves / 256-thr block) ----
  k_stem<<<(B_ * (C_ / 16) * (N_ / 64)) / 8, 256, 0, stream>>>(w1h, inT, s1, sh1, x, xcatT);
  k_qk  <<<(B_ * (CIN_ / 16) * (N_ / 64)) / 8, 256, 0, stream>>>(wqkh, xcatT, xqkT);
  k_v   <<<(B_ * (C_ / 16) * (N_ / 64)) / 8, 256, 0, stream>>>(vwh, xcatT, v_b, xv);
  k_attn<<<B_ * (N_ / 16), 256, 16 * N_ * sizeof(float), stream>>>(xqkT, rel_pos, attnT, colsum);
  k_xr  <<<B_ * 2 * (N_ / 64), 256, 0, stream>>>(xv, attnT, colsum, x, tT);
  k_trans<<<(B_ * (C_ / 16) * (N_ / 64)) / 8, 256, 0, stream>>>(twh, tT, trans_b, san, shan, x, xcatT);
  k_fuse<<<(B_ * (C_ / 16) * (N_ / 64)) / 8, 256, 0, stream>>>(fwh, xcatT, sf, shf, out);
}
